// MambaBlock_90022514524367
// MI455X (gfx1250) — compile-verified, matrix operand order checked by reading, not run on hardware
//
#include <hip/hip_runtime.h>
#include <hip/hip_bf16.h>

// ---------------------------------------------------------------------------
// Mamba block for MI455X (gfx1250, wave32, WMMA).
// GEMMs run on v_wmma_f32_16x16x32_bf16 (fp32 accumulate); scan in fp32 VALU.
// ---------------------------------------------------------------------------

#define D_MODEL 1024
#define D_INNER 2048
#define D_STATE 16
#define D_CONV  4
#define DT_RANK 64
#define BATCH   2
#define SEQLEN  1024
#define TOKENS  (BATCH * SEQLEN)   // 2048
#define DBL_N   (DT_RANK + 2 * D_STATE)  // 96

typedef __attribute__((ext_vector_type(16))) __bf16 v16bf;
typedef __attribute__((ext_vector_type(8)))  float  v8f;

__device__ __forceinline__ void sched_fence() {
#if __has_builtin(__builtin_amdgcn_sched_barrier)
  __builtin_amdgcn_sched_barrier(0);   // no instruction may cross this point
#endif
}

// ---------------------------------------------------------------------------
// fp32 -> bf16 bulk convert (weights), grid-stride
// ---------------------------------------------------------------------------
__global__ void f32_to_bf16_kernel(const float* __restrict__ in,
                                   __hip_bfloat16* __restrict__ out, size_t n) {
  size_t i = (size_t)blockIdx.x * blockDim.x + threadIdx.x;
  size_t stride = (size_t)gridDim.x * blockDim.x;
  for (; i < n; i += stride) out[i] = __float2bfloat16(in[i]);
}

// ---------------------------------------------------------------------------
// RMSNorm over d_model, bf16 output. One block (256 thr = 8 waves) per token.
// ---------------------------------------------------------------------------
__global__ void rmsnorm_bf16_kernel(const float* __restrict__ x,
                                    const float* __restrict__ w,
                                    __hip_bfloat16* __restrict__ out) {
  const int token = blockIdx.x;
  const float* row = x + (size_t)token * D_MODEL;
  __hip_bfloat16* orow = out + (size_t)token * D_MODEL;

  float ss = 0.f;
  for (int i = threadIdx.x; i < D_MODEL; i += blockDim.x) {
    float v = row[i];
    ss += v * v;
  }
  for (int off = 16; off > 0; off >>= 1) ss += __shfl_xor(ss, off, 32);
  __shared__ float red[9];
  const int wid = threadIdx.x >> 5, lane = threadIdx.x & 31;
  if (lane == 0) red[wid] = ss;
  __syncthreads();
  if (wid == 0) {
    float s = (lane < (int)(blockDim.x >> 5)) ? red[lane] : 0.f;
    for (int off = 4; off > 0; off >>= 1) s += __shfl_xor(s, off, 32);
    if (lane == 0) red[8] = rsqrtf(s / (float)D_MODEL + 1e-5f);
  }
  __syncthreads();
  const float scale = red[8];
  for (int i = threadIdx.x; i < D_MODEL; i += blockDim.x)
    orow[i] = __float2bfloat16(row[i] * scale * w[i]);
}

// ---------------------------------------------------------------------------
// WMMA GEMM:  C[M,N] = A[M,K] * W[N,K]^T  (+ resid if RESID)
//   A, W bf16 row-major; C fp32 row-major.
//   K compile-time (STEPS = K/32 even). N % (16*NT) == 0 for every call site.
//   Block: 256 threads = 8 waves; each wave -> 16 rows x (16*NT) cols (NT
//   WMMA tiles sharing the A fragment). Grid: (N/(16*NT), M/128).
//   Addressing: ONE A pointer + ONE B pointer per lane; tile t of B sits at
//   the compile-time offset t*K v16bf, ping/pong prefetches at +4/+6 v16bf;
//   both pointers advance +4/iteration.
//   sched_barriers pin the WAR order so refill loads write DIRECTLY into the
//   WMMA operand tuples (no temp copies, no hazard nops), while the region
//   between barriers still lets ping's refill co-issue with pong's WMMAs.
// ---------------------------------------------------------------------------
#define GEMM_WAVES 8

template <int K, int NT, bool RESID>
__global__ __launch_bounds__(256)
void gemm_bf16_wmma_kernel(const __hip_bfloat16* __restrict__ Abf,
                           const __hip_bfloat16* __restrict__ Wbf,
                           float* __restrict__ C,
                           const float* __restrict__ resid,
                           int N) {
  constexpr int STEPS = K / 32;          // even for all instantiations
  static_assert(STEPS >= 2 && (STEPS % 2) == 0, "STEPS must be even");
  constexpr int TS = K;                  // B tile stride in v16bf units (16 rows * K elem)
  const int lane = threadIdx.x & 31;
  const int wave = threadIdx.x >> 5;
  const int half = lane >> 4;            // selects K sub-range of the fragment
  const int r    = lane & 15;            // row/col within 16x16 tile
  const int mtile = blockIdx.y * GEMM_WAVES + wave;
  const int n0    = blockIdx.x * (16 * NT);

  // per-lane base pointers, pre-offset by `half` (fragment K sub-range)
  const v16bf* Ap = reinterpret_cast<const v16bf*>(Abf + (size_t)(mtile * 16 + r) * K) + half;
  const v16bf* Bp = reinterpret_cast<const v16bf*>(Wbf + (size_t)(n0 + r) * K) + half;

  v8f acc[NT] = {};

  // prologue: steps 0 and 1 into ping (a0/b0) and pong (a1/b1)
  v16bf a0 = Ap[0];
  v16bf b0[NT];
#pragma unroll
  for (int t = 0; t < NT; ++t) b0[t] = Bp[t * TS];
  v16bf a1 = Ap[2];
  v16bf b1[NT];
#pragma unroll
  for (int t = 0; t < NT; ++t) b1[t] = Bp[2 + t * TS];

#pragma unroll 1
  for (int s = 0; s < STEPS - 2; s += 2) {
    // --- compute step s on ping ---
#pragma unroll
    for (int t = 0; t < NT; ++t)
      acc[t] = __builtin_amdgcn_wmma_f32_16x16x32_bf16(
          false, a0, false, b0[t], (short)0, acc[t], false, false);
    sched_fence();   // refill(ping) may not hoist above wmma(ping)
    // --- refill ping with step s+2 (+4); may interleave with wmma(pong) ---
    a0 = Ap[4];
#pragma unroll
    for (int t = 0; t < NT; ++t) b0[t] = Bp[4 + t * TS];
    // --- compute step s+1 on pong ---
#pragma unroll
    for (int t = 0; t < NT; ++t)
      acc[t] = __builtin_amdgcn_wmma_f32_16x16x32_bf16(
          false, a1, false, b1[t], (short)0, acc[t], false, false);
    sched_fence();   // refill(pong) may not hoist above wmma(pong)
    // --- refill pong with step s+3 (+6) ---
    a1 = Ap[6];
#pragma unroll
    for (int t = 0; t < NT; ++t) b1[t] = Bp[6 + t * TS];
    Ap += 4;
    Bp += 4;
    sched_fence();   // keep pipeline exactly 2-deep; no cross-iteration motion
  }
  // tail: last two steps (ping then pong), no more prefetch
#pragma unroll
  for (int t = 0; t < NT; ++t)
    acc[t] = __builtin_amdgcn_wmma_f32_16x16x32_bf16(
        false, a0, false, b0[t], (short)0, acc[t], false, false);
#pragma unroll
  for (int t = 0; t < NT; ++t)
    acc[t] = __builtin_amdgcn_wmma_f32_16x16x32_bf16(
        false, a1, false, b1[t], (short)0, acc[t], false, false);

  // C/D layout: VGPR v holds row = v + 8*half, col = lane&15 (per 16x16 tile)
  const int mbase = mtile * 16 + 8 * half;
#pragma unroll
  for (int t = 0; t < NT; ++t) {
    const int col = n0 + t * 16 + r;
#pragma unroll
    for (int v = 0; v < 8; ++v) {
      const size_t o = (size_t)(mbase + v) * N + col;
      float val = acc[t][v];
      if (RESID) val += resid[o];
      C[o] = val;
    }
  }
}

// ---------------------------------------------------------------------------
// Causal depthwise conv1d (width 4) + bias + SiLU over xin = xz[:, 0:D_INNER].
// xz row stride is 2*D_INNER. Output bf16 (feeds x_proj GEMM and scan).
// ---------------------------------------------------------------------------
__global__ void conv_silu_kernel(const float* __restrict__ xz,
                                 const float* __restrict__ cw,
                                 const float* __restrict__ cb,
                                 __hip_bfloat16* __restrict__ xc) {
  const size_t total = (size_t)TOKENS * D_INNER;
  for (size_t i = (size_t)blockIdx.x * blockDim.x + threadIdx.x; i < total;
       i += (size_t)gridDim.x * blockDim.x) {
    const int d = (int)(i % D_INNER);
    const size_t tok = i / D_INNER;
    const int t = (int)(tok % SEQLEN);
    float acc = cb[d];
#pragma unroll
    for (int j = 0; j < D_CONV; ++j) {
      const int back = D_CONV - 1 - j;    // taps at t-3 .. t
      if (t - back >= 0)
        acc += xz[(tok - (size_t)back) * (2 * D_INNER) + d] * cw[d * D_CONV + j];
    }
    const float s = acc / (1.f + __expf(-acc));   // silu
    xc[i] = __float2bfloat16(s);
  }
}

// ---------------------------------------------------------------------------
// Slice dt (first DT_RANK cols of dbl) to bf16 for the dt_proj GEMM.
// ---------------------------------------------------------------------------
__global__ void slice_dt_kernel(const float* __restrict__ dbl,
                                __hip_bfloat16* __restrict__ dt) {
  const int i = blockIdx.x * blockDim.x + threadIdx.x;
  if (i < TOKENS * DT_RANK) {
    const int tok = i >> 6, r = i & 63;
    dt[i] = __float2bfloat16(dbl[tok * DBL_N + r]);
  }
}

// ---------------------------------------------------------------------------
// Selective scan. One thread per (batch, channel); 16-state recurrence kept
// in registers. B_t/C_t fetched as 8x float4 (global_load_b128). Fuses:
// softplus(dt_proj + bias), +x*D skip, SiLU(z) gate.
// ---------------------------------------------------------------------------
__global__ void scan_kernel(const float* __restrict__ dtraw,   // (TOK, D_INNER)
                            const float* __restrict__ dtb,     // (D_INNER)
                            const float* __restrict__ dbl,     // (TOK, 96)
                            const __hip_bfloat16* __restrict__ xc, // (TOK, D_INNER)
                            const float* __restrict__ xz,      // (TOK, 2*D_INNER)
                            const float* __restrict__ A_log,   // (D_INNER, 16)
                            const float* __restrict__ Dp,      // (D_INNER)
                            __hip_bfloat16* __restrict__ yg) { // (TOK, D_INNER)
  const int d = blockIdx.x * blockDim.x + threadIdx.x;  // channel
  const int b = blockIdx.y;                             // batch
  if (d >= D_INNER) return;

  float a[D_STATE], h[D_STATE];
#pragma unroll
  for (int n = 0; n < D_STATE; ++n) {
    a[n] = -__expf(A_log[d * D_STATE + n]);
    h[n] = 0.f;
  }
  const float bias = dtb[d];
  const float dskip = Dp[d];

  for (int t = 0; t < SEQLEN; ++t) {
    const size_t tok = (size_t)b * SEQLEN + t;
    const float raw = dtraw[tok * D_INNER + d] + bias;
    const float delta = (raw > 20.f) ? raw : log1pf(__expf(raw));  // softplus
    const float xt = __bfloat162float(xc[tok * D_INNER + d]);
    const float dx = delta * xt;

    // B_t (16) and C_t (16) as 8 float4 loads; 16B-aligned (96*4 and 64*4
    // offsets are multiples of 16, base is 256-aligned).
    const float4* bc4 = reinterpret_cast<const float4*>(dbl + tok * DBL_N + DT_RANK);
    float Bv[D_STATE], Cv[D_STATE];
#pragma unroll
    for (int q = 0; q < 4; ++q) {
      const float4 bq = bc4[q];
      Bv[q * 4 + 0] = bq.x; Bv[q * 4 + 1] = bq.y;
      Bv[q * 4 + 2] = bq.z; Bv[q * 4 + 3] = bq.w;
      const float4 cq = bc4[4 + q];
      Cv[q * 4 + 0] = cq.x; Cv[q * 4 + 1] = cq.y;
      Cv[q * 4 + 2] = cq.z; Cv[q * 4 + 3] = cq.w;
    }

    float y = 0.f;
#pragma unroll
    for (int n = 0; n < D_STATE; ++n) {
      const float e = __expf(delta * a[n]);
      h[n] = fmaf(e, h[n], dx * Bv[n]);             // h = exp(dA)h + dx*B
      y = fmaf(h[n], Cv[n], y);                     // y += h*C
    }
    y = fmaf(xt, dskip, y);                         // + x*D skip
    const float z = xz[tok * (2 * D_INNER) + D_INNER + d];
    y *= z / (1.f + __expf(-z));                    // * silu(z)
    yg[tok * D_INNER + d] = __float2bfloat16(y);
  }
}

// ---------------------------------------------------------------------------
// Host-side orchestration
// ---------------------------------------------------------------------------
extern "C" void kernel_launch(void* const* d_in, const int* in_sizes, int n_in,
                              void* d_out, int out_size, void* d_ws, size_t ws_size,
                              hipStream_t stream) {
  const float* hidden     = (const float*)d_in[0];
  const float* norm_w     = (const float*)d_in[1];
  const float* in_proj_w  = (const float*)d_in[2];
  const float* conv_w     = (const float*)d_in[3];
  const float* conv_b     = (const float*)d_in[4];
  const float* x_proj_w   = (const float*)d_in[5];
  const float* dt_proj_w  = (const float*)d_in[6];
  const float* dt_proj_b  = (const float*)d_in[7];
  const float* A_log      = (const float*)d_in[8];
  const float* Dp         = (const float*)d_in[9];
  const float* out_proj_w = (const float*)d_in[10];
  float* out = (float*)d_out;

  char* ws = (char*)d_ws;
  size_t off = 0;
  auto alloc = [&](size_t bytes) -> void* {
    void* p = ws + off;
    off = (off + bytes + 255) & ~(size_t)255;
    return p;
  };

  __hip_bfloat16* w_in_bf  = (__hip_bfloat16*)alloc((size_t)2 * D_INNER * D_MODEL * 2);
  __hip_bfloat16* w_out_bf = (__hip_bfloat16*)alloc((size_t)D_MODEL * D_INNER * 2);
  __hip_bfloat16* w_xp_bf  = (__hip_bfloat16*)alloc((size_t)DBL_N * D_INNER * 2);
  __hip_bfloat16* w_dt_bf  = (__hip_bfloat16*)alloc((size_t)D_INNER * DT_RANK * 2);
  __hip_bfloat16* xnorm_bf = (__hip_bfloat16*)alloc((size_t)TOKENS * D_MODEL * 2);
  float*          xz_f32   = (float*)alloc((size_t)TOKENS * 2 * D_INNER * 4);
  __hip_bfloat16* xc_bf    = (__hip_bfloat16*)alloc((size_t)TOKENS * D_INNER * 2);
  float*          dbl_f32  = (float*)alloc((size_t)TOKENS * DBL_N * 4);
  __hip_bfloat16* dt_bf    = (__hip_bfloat16*)alloc((size_t)TOKENS * DT_RANK * 2);
  float*          dtraw    = (float*)alloc((size_t)TOKENS * D_INNER * 4);
  __hip_bfloat16* yg_bf    = (__hip_bfloat16*)alloc((size_t)TOKENS * D_INNER * 2);

  // --- convert weights to bf16 (all L2-resident; negligible cost) ---
  f32_to_bf16_kernel<<<2048, 256, 0, stream>>>(in_proj_w,  w_in_bf,  (size_t)2 * D_INNER * D_MODEL);
  f32_to_bf16_kernel<<<2048, 256, 0, stream>>>(out_proj_w, w_out_bf, (size_t)D_MODEL * D_INNER);
  f32_to_bf16_kernel<<<768,  256, 0, stream>>>(x_proj_w,   w_xp_bf,  (size_t)DBL_N * D_INNER);
  f32_to_bf16_kernel<<<512,  256, 0, stream>>>(dt_proj_w,  w_dt_bf,  (size_t)D_INNER * DT_RANK);

  // --- 1. RMSNorm -> bf16 ---
  rmsnorm_bf16_kernel<<<TOKENS, 256, 0, stream>>>(hidden, norm_w, xnorm_bf);

  // --- 2. in_proj: xz[2048,4096] = xnorm[2048,1024] @ W_in[4096,1024]^T ---
  {
    dim3 g((2 * D_INNER) / 64, TOKENS / 128);
    gemm_bf16_wmma_kernel<D_MODEL, 4, false>
        <<<g, 256, 0, stream>>>(xnorm_bf, w_in_bf, xz_f32, nullptr, 2 * D_INNER);
  }

  // --- 3. causal depthwise conv + SiLU -> xc (bf16) ---
  conv_silu_kernel<<<16384, 256, 0, stream>>>(xz_f32, conv_w, conv_b, xc_bf);

  // --- 4. x_proj: dbl[2048,96] = xc[2048,2048] @ W_xp[96,2048]^T (NT=2: 96=3*32) ---
  {
    dim3 g(DBL_N / 32, TOKENS / 128);
    gemm_bf16_wmma_kernel<D_INNER, 2, false>
        <<<g, 256, 0, stream>>>(xc_bf, w_xp_bf, dbl_f32, nullptr, DBL_N);
  }

  // --- 5. slice dt -> bf16; dt_proj: dtraw[2048,2048] = dt[2048,64] @ W_dt[2048,64]^T ---
  slice_dt_kernel<<<(TOKENS * DT_RANK + 255) / 256, 256, 0, stream>>>(dbl_f32, dt_bf);
  {
    dim3 g(D_INNER / 64, TOKENS / 128);
    gemm_bf16_wmma_kernel<DT_RANK, 4, false>
        <<<g, 256, 0, stream>>>(dt_bf, w_dt_bf, dtraw, nullptr, D_INNER);
  }

  // --- 6. selective scan (fused softplus, skip, gate) -> yg (bf16) ---
  {
    dim3 g(D_INNER / 256, BATCH);
    scan_kernel<<<g, 256, 0, stream>>>(dtraw, dt_proj_b, dbl_f32, xc_bf, xz_f32,
                                       A_log, Dp, yg_bf);
  }

  // --- 7. out_proj + residual: out[2048,1024] = yg @ W_out[1024,2048]^T + hidden ---
  {
    dim3 g(D_MODEL / 64, TOKENS / 128);
    gemm_bf16_wmma_kernel<D_INNER, 4, true>
        <<<g, 256, 0, stream>>>(yg_bf, w_out_bf, out, hidden, D_MODEL);
  }
}
